// GATConvEdges_21706764714723
// MI455X (gfx1250) — matrix operand order repaired and checked by reading, not compiled, not run on hardware
//
#include <hip/hip_runtime.h>

// Reference returns only: pooled = leaky(edge_attr@Wp1^T + bp1)@Wp2^T + bp2  ([E,64], f32).
// HBM-bound (256 MB @ 23.3 TB/s ~= 11 us); f16 WMMA w/ fp32 accum makes compute free.
// This version double-buffers edge_attr tiles with GLOBAL_LOAD_ASYNC_TO_LDS_B128
// (ASYNCcnt) so HBM streaming overlaps the WMMA pipeline.

typedef _Float16 v16h  __attribute__((ext_vector_type(16)));
typedef _Float16 half8 __attribute__((ext_vector_type(8)));
typedef float    v8f   __attribute__((ext_vector_type(8)));
typedef float    f8v   __attribute__((ext_vector_type(8)));

#define WSTRIDE 80    // halfs per weight/hidden row (160 B, 16-B aligned)
#define TSTRIDE 72    // floats per tile row (288 B, 16-B aligned)
#define ROWS    128   // rows per block tile (8 waves x 16 rows)
#define THREADS 256
#define LSLOPE  0.01f

__global__ __launch_bounds__(THREADS)
void pooled_mlp_kernel(const float* __restrict__ attr,
                       const float* __restrict__ Wp1, const float* __restrict__ bp1,
                       const float* __restrict__ Wp2, const float* __restrict__ bp2,
                       float* __restrict__ out, int E)
{
    __shared__ float    tileS[2][ROWS * TSTRIDE];   // double-buffered f32 edge_attr tiles
    __shared__ _Float16 w1s[64 * WSTRIDE];
    __shared__ _Float16 w2s[64 * WSTRIDE];
    __shared__ _Float16 hidS[ROWS * WSTRIDE];       // f16 hidden (layer1 -> layer2)
    __shared__ float    b1s[64], b2s[64];

    const int tid   = threadIdx.x;
    const int lane  = tid & 31;
    const int wave  = tid >> 5;
    const int lrow  = lane & 15;
    const int lhalf = lane >> 4;
    const int stride = (int)gridDim.x * ROWS;
    const int tile0  = (int)blockIdx.x * ROWS;

    // ---- async copy of one 128x64 f32 tile into LDS buffer `buf` (8 x b128 per thread)
    auto issue_copy = [&](int buf, int tileBase) {
        #pragma unroll
        for (int j = 0; j < 8; ++j) {
            const int i  = tid + j * THREADS;       // 0..2047 float4 slots
            const int r  = i >> 4, c4 = i & 15;
            int row = tileBase + r;
            if (row > E - 1) row = E - 1;           // clamp tail; stores predicated later
            const float4* g = (const float4*)attr + (size_t)row * 16 + c4;
            const unsigned lds = (unsigned)(uintptr_t)&tileS[buf][r * TSTRIDE + c4 * 4];
            asm volatile("global_load_async_to_lds_b128 %0, %1, off"
                         :: "v"(lds), "v"((unsigned long long)(uintptr_t)g)
                         : "memory");
        }
    };

    // ---- kick off the first tile copy immediately, then stage weights behind it
    if (tile0 < E) issue_copy(0, tile0);

    for (int idx = tid; idx < 64 * 64; idx += THREADS) {
        const int n = idx >> 6, k = idx & 63;
        w1s[n * WSTRIDE + k] = (_Float16)Wp1[idx];
        w2s[n * WSTRIDE + k] = (_Float16)Wp2[idx];
    }
    if (tid < 64) { b1s[tid] = bp1[tid]; b2s[tid] = bp2[tid]; }
    __syncthreads();

    // ---- register-resident B fragments for both layers (constant across tiles)
    // lane holds column n = 16t + lrow; e=0..7 -> K = 32s + 8*lhalf + e ; e=8..15 -> +16.
    v16h B1[8], B2[8];
    #pragma unroll
    for (int t = 0; t < 4; ++t) {
        #pragma unroll
        for (int s = 0; s < 2; ++s) {
            const int n  = 16 * t + lrow;
            const int kb = 32 * s + 8 * lhalf;
            const half8 c0 = *(const half8*)&w1s[n * WSTRIDE + kb];
            const half8 c1 = *(const half8*)&w1s[n * WSTRIDE + kb + 16];
            const half8 d0 = *(const half8*)&w2s[n * WSTRIDE + kb];
            const half8 d1 = *(const half8*)&w2s[n * WSTRIDE + kb + 16];
            v16h b, d;
            #pragma unroll
            for (int e = 0; e < 8; ++e) {
                b[e] = c0[e]; b[8 + e] = c1[e];
                d[e] = d0[e]; d[8 + e] = d1[e];
            }
            B1[t * 2 + s] = b;
            B2[t * 2 + s] = d;
        }
    }

    int buf = 0;
    for (int tile = tile0; tile < E; tile += stride, buf ^= 1) {
        // ---- pipeline: stream tile i+1 into the other buffer while computing tile i
        const int nt = tile + stride;               // block-uniform
        const bool haveNext = nt < E;
        if (haveNext) issue_copy(buf ^ 1, nt);
        const int nt2 = tile + 2 * stride;
        if (nt2 < E) __builtin_prefetch(attr + (size_t)nt2 * 64 + tid * 16, 0, 1);

        // async loads complete in order: <=8 outstanding means copy(i) has landed
        if (haveNext) asm volatile("s_wait_asynccnt 0x8" ::: "memory");
        else          asm volatile("s_wait_asynccnt 0x0" ::: "memory");
        __syncthreads();

        const float* myRows = &tileS[buf][(wave * 16 + lrow) * TSTRIDE];

        // ---- layer 1: hidden = leaky(attr @ W1^T + b1), A built from f32 LDS (cvt here)
        v8f acc[4] = {v8f{}, v8f{}, v8f{}, v8f{}};
        #pragma unroll
        for (int s = 0; s < 2; ++s) {
            // A layout: e=0..7 -> K = 32s + 8*lhalf + e ; e=8..15 -> +16
            const float* p = myRows + 32 * s + 8 * lhalf;
            const f8v c0 = *(const f8v*)p;
            const f8v c1 = *(const f8v*)(p + 16);
            v16h a;
            #pragma unroll
            for (int e = 0; e < 8; ++e) { a[e] = (_Float16)c0[e]; a[8 + e] = (_Float16)c1[e]; }
            #pragma unroll
            for (int t = 0; t < 4; ++t)
                acc[t] = __builtin_amdgcn_wmma_f32_16x16x32_f16(
                    false, a, false, B1[t * 2 + s], (short)0, acc[t], false, false);
        }

        // ---- epilogue 1: bias + leaky -> f16 hidden rows (wave-private, DS in-order)
        #pragma unroll
        for (int t = 0; t < 4; ++t) {
            const int n = 16 * t + lrow;
            const float b = b1s[n];
            #pragma unroll
            for (int r = 0; r < 8; ++r) {
                float v = acc[t][r] + b;
                v = v > 0.f ? v : LSLOPE * v;
                const int m = r + 8 * lhalf;
                hidS[(wave * 16 + m) * WSTRIDE + n] = (_Float16)v;
            }
        }

        // ---- layer 2: out = hidden @ W2^T + b2
        const _Float16* myHid = &hidS[(wave * 16 + lrow) * WSTRIDE];
        v8f acc2[4] = {v8f{}, v8f{}, v8f{}, v8f{}};
        #pragma unroll
        for (int s = 0; s < 2; ++s) {
            const _Float16* p = myHid + 32 * s + 8 * lhalf;
            const half8 c0 = *(const half8*)p;
            const half8 c1 = *(const half8*)(p + 16);
            v16h a;
            #pragma unroll
            for (int e = 0; e < 8; ++e) { a[e] = c0[e]; a[8 + e] = c1[e]; }
            #pragma unroll
            for (int t = 0; t < 4; ++t)
                acc2[t] = __builtin_amdgcn_wmma_f32_16x16x32_f16(
                    false, a, false, B2[t * 2 + s], (short)0, acc2[t], false, false);
        }

        // ---- epilogue 2: bias + global store (predicated only at the tail)
        #pragma unroll
        for (int t = 0; t < 4; ++t) {
            const int n = 16 * t + lrow;
            const float b = b2s[n];
            #pragma unroll
            for (int r = 0; r < 8; ++r) {
                const int m = r + 8 * lhalf;
                const int grow = tile + wave * 16 + m;
                if (grow < E)
                    out[(size_t)grow * 64 + n] = acc2[t][r] + b;
            }
        }
        __syncthreads();   // all waves done with tileS[buf] before copy(i+2) targets it
    }
}

extern "C" void kernel_launch(void* const* d_in, const int* in_sizes, int n_in,
                              void* d_out, int out_size, void* d_ws, size_t ws_size,
                              hipStream_t stream) {
    // setup_inputs() order: x(0), edge_index(1), edge_attr(2), x_0(3),
    // Wa_in(4), ba_in(5), Wa_out(6), ba_out(7), Wm_in(8), bm_in(9),
    // Wm_out(10), bm_out(11), Wp1(12), bp1(13), Wp2(14), bp2(15)
    const float* attr = (const float*)d_in[2];
    const float* Wp1  = (const float*)d_in[12];
    const float* bp1  = (const float*)d_in[13];
    const float* Wp2  = (const float*)d_in[14];
    const float* bp2  = (const float*)d_in[15];
    float* out = (float*)d_out;

    const int E = in_sizes[2] / 64;           // 500000
    const int tiles = (E + ROWS - 1) / ROWS;  // 3907
    const int grid = tiles < 1024 ? tiles : 1024;
    pooled_mlp_kernel<<<grid, THREADS, 0, stream>>>(attr, Wp1, bp1, Wp2, bp2, out, E);
}
